// TimesNet_11854109737595
// MI455X (gfx1250) — compile-verified, matrix operand order checked by reading, not run on hardware
//
#include <hip/hip_runtime.h>
#include <hip/hip_bf16.h>
#include <math.h>

// ---------------------------------------------------------------------------
// Problem constants (from reference)
// ---------------------------------------------------------------------------
#define BB      16
#define SS      960
#define DIN     7
#define HID     512
#define HEADS   8
#define HEAD_DIM 64
#define LAYERS  4
#define FFN     2048
#define OUT_LEN 96
#define TOKENS  (BB * SS)          // 15360

typedef _Float16 v8h  __attribute__((ext_vector_type(8)));
typedef _Float16 v16h __attribute__((ext_vector_type(16)));
typedef float    v8f  __attribute__((ext_vector_type(8)));
typedef int      b128_t __attribute__((vector_size(16)));

// ---------------------------------------------------------------------------
// gfx1250 async global->LDS copy (ASYNCcnt path) with safe fallback
// ---------------------------------------------------------------------------
#if defined(__HIP_DEVICE_COMPILE__) && \
    __has_builtin(__builtin_amdgcn_global_load_async_to_lds_b128)
#define USE_ASYNC_LDS 1
#else
#define USE_ASYNC_LDS 0
#endif

__device__ __forceinline__ void async_cp16(const _Float16* g, _Float16* l) {
#if USE_ASYNC_LDS
    __builtin_amdgcn_global_load_async_to_lds_b128(
        (__attribute__((address_space(1))) b128_t*)(g),
        (__attribute__((address_space(3))) b128_t*)(l), 0, 0);
#else
    *(v8h*)l = *(const v8h*)g;
#endif
}

__device__ __forceinline__ void async_cp_wait() {
#if USE_ASYNC_LDS
#if __has_builtin(__builtin_amdgcn_s_wait_asynccnt)
    __builtin_amdgcn_s_wait_asynccnt(0);
#else
    asm volatile("s_wait_asynccnt 0" ::: "memory");
#endif
#endif
}

// ---------------------------------------------------------------------------
// f32 -> f16 convert (grid-strided)
// ---------------------------------------------------------------------------
__global__ void convert_f16_kernel(const float* __restrict__ src,
                                   _Float16* __restrict__ dst, int n) {
    for (int i = blockIdx.x * blockDim.x + threadIdx.x; i < n;
         i += gridDim.x * blockDim.x)
        dst[i] = (_Float16)src[i];
}

// ---------------------------------------------------------------------------
// Input projection: h0[t,n] = sum_k x[t,k]*W[n,k] + b[n]   (K = 7, trivial)
// ---------------------------------------------------------------------------
__global__ void input_proj_kernel(const float* __restrict__ x,
                                  const float* __restrict__ w,
                                  const float* __restrict__ b,
                                  float* __restrict__ h0) {
    int gid = blockIdx.x * blockDim.x + threadIdx.x;
    if (gid >= TOKENS * HID) return;
    int t = gid >> 9;          // /512
    int n = gid & 511;
    const float* xr = x + t * DIN;
    const float* wr = w + n * DIN;
    float s = b[n];
#pragma unroll
    for (int k = 0; k < DIN; ++k) s += xr[k] * wr[k];
    h0[gid] = s;
}

// ---------------------------------------------------------------------------
// Branch init: x = h0 ; x16 = f16(h0)
// ---------------------------------------------------------------------------
__global__ void branch_init_kernel(const float* __restrict__ h0,
                                   float* __restrict__ x,
                                   _Float16* __restrict__ x16, int n) {
    for (int i = blockIdx.x * blockDim.x + threadIdx.x; i < n;
         i += gridDim.x * blockDim.x) {
        float v = h0[i];
        x[i] = v;
        x16[i] = (_Float16)v;
    }
}

// ---------------------------------------------------------------------------
// Branch accumulate: acc = x (first) or acc += x
// ---------------------------------------------------------------------------
__global__ void accum_kernel(const float* __restrict__ x,
                             float* __restrict__ acc, int n, int first) {
    for (int i = blockIdx.x * blockDim.x + threadIdx.x; i < n;
         i += gridDim.x * blockDim.x)
        acc[i] = first ? x[i] : acc[i] + x[i];
}

// ---------------------------------------------------------------------------
// WMMA GEMM:  C[M,N] = A[M,K] (f16, row-major) * W[N,K]^T (f16, row-major)
//             + bias, optional exact GELU, output f32 or f16.
// Block tile 128x128, BK=32, 256 threads = 8 waves (4 in M x 2 in N);
// wave tile 32x64 = 2x4 WMMA tiles -> 8 v_wmma per K-step per wave.
// Double-buffered LDS, async global->LDS fills, prefetch 2 K-steps ahead.
// ---------------------------------------------------------------------------
#define BM 128
#define BN 128
#define BK 32
#define LDS_STRIDE 40   // halves; pads banks, keeps 16B alignment of frags

template <bool GELU, bool OUT16>
__global__ __launch_bounds__(256)
void gemm_wmma_kernel(const _Float16* __restrict__ A,
                      const _Float16* __restrict__ W,
                      const float* __restrict__ bias,
                      void* __restrict__ outp,
                      int M, int N, int K) {
    __shared__ __attribute__((aligned(16))) _Float16 As[2][BM * LDS_STRIDE];
    __shared__ __attribute__((aligned(16))) _Float16 Bs[2][BN * LDS_STRIDE];

    const int tid  = threadIdx.x;
    const int lane = tid & 31;
    const int wid  = tid >> 5;
    const int wm   = (wid & 3) * 32;   // wave row offset inside block tile
    const int wn   = (wid >> 2) * 64;  // wave col offset inside block tile
    const int l16  = lane & 15;
    const int kg   = (lane < 16) ? 0 : 8;   // ISA 16-bit A/B lane K grouping

    const int bm = blockIdx.x * BM;
    const int bn = blockIdx.y * BN;

    // tile-fill assignment: both tiles are 128 rows x 32 halves
    const int ldRow = tid >> 1;           // 0..127
    const int ldCol = (tid & 1) * 16;     // 0 or 16 halves

    const _Float16* gaBase = A + (size_t)(bm + ldRow) * K + ldCol;
    const _Float16* gbBase = W + (size_t)(bn + ldRow) * K + ldCol;

    v8f acc[2][4];
#pragma unroll
    for (int mi = 0; mi < 2; ++mi)
#pragma unroll
        for (int ni = 0; ni < 4; ++ni) acc[mi][ni] = (v8f)0.0f;

    // ---- fill helper (async when available) ----
    auto fill = [&](int bufi, int k0) {
        const _Float16* ga = gaBase + k0;
        const _Float16* gb = gbBase + k0;
        _Float16* la = &As[bufi][ldRow * LDS_STRIDE + ldCol];
        _Float16* lb = &Bs[bufi][ldRow * LDS_STRIDE + ldCol];
        async_cp16(ga,     la);
        async_cp16(ga + 8, la + 8);
        async_cp16(gb,     lb);
        async_cp16(gb + 8, lb + 8);
        // stream hint two K-steps ahead (lowers to global_prefetch_b8)
        if (k0 + 2 * BK < K) {
            __builtin_prefetch(ga + 2 * BK, 0, 1);
            __builtin_prefetch(gb + 2 * BK, 0, 1);
        }
    };

    // ---- prologue: fill buffer 0 ----
    fill(0, 0);
    async_cp_wait();
    __syncthreads();

    int buf = 0;
    for (int k0 = 0; k0 < K; k0 += BK) {
        const int nbuf = buf ^ 1;
        if (k0 + BK < K) fill(nbuf, k0 + BK);   // overlap with compute

        // ---- build fragments from LDS & multiply ----
        v16h afrag[2], bfrag[4];
#pragma unroll
        for (int mi = 0; mi < 2; ++mi) {
            const _Float16* p = &As[buf][(wm + mi * 16 + l16) * LDS_STRIDE + kg];
            ((v8h*)&afrag[mi])[0] = *(const v8h*)p;
            ((v8h*)&afrag[mi])[1] = *(const v8h*)(p + 16);
        }
#pragma unroll
        for (int ni = 0; ni < 4; ++ni) {
            const _Float16* p = &Bs[buf][(wn + ni * 16 + l16) * LDS_STRIDE + kg];
            ((v8h*)&bfrag[ni])[0] = *(const v8h*)p;
            ((v8h*)&bfrag[ni])[1] = *(const v8h*)(p + 16);
        }
#pragma unroll
        for (int mi = 0; mi < 2; ++mi)
#pragma unroll
            for (int ni = 0; ni < 4; ++ni)
                acc[mi][ni] = __builtin_amdgcn_wmma_f32_16x16x32_f16(
                    false, afrag[mi], false, bfrag[ni],
                    (short)0, acc[mi][ni], false, false);

        async_cp_wait();    // next buffer landed
        __syncthreads();    // everyone done reading buf / filling nbuf
        buf = nbuf;
    }

    // ---- epilogue: bias (+GELU), store f32 or f16 ----
    float* outF = (float*)outp;
    _Float16* outH = (_Float16*)outp;
    const int rowHi = (lane >> 4) << 3;   // 0 or 8 (C/D layout)
#pragma unroll
    for (int mi = 0; mi < 2; ++mi) {
#pragma unroll
        for (int ni = 0; ni < 4; ++ni) {
            const int col = bn + wn + ni * 16 + l16;
            const float bv = bias[col];
#pragma unroll
            for (int i = 0; i < 8; ++i) {
                const int row = bm + wm + mi * 16 + i + rowHi;
                float v = acc[mi][ni][i] + bv;
                if (GELU)
                    v = 0.5f * v * (1.0f + erff(v * 0.70710678118654752f));
                const size_t idx = (size_t)row * N + col;
                if (OUT16) outH[idx] = (_Float16)v;
                else       outF[idx] = v;
            }
        }
    }
}

// ---------------------------------------------------------------------------
// Attention: one block per (sequence, head). p in {24,48,96}.
// qkv f16 [TOKENS,1536] (q|k|v each 512). Writes attn_out f16 [TOKENS,512].
// ---------------------------------------------------------------------------
__global__ __launch_bounds__(128)
void attention_kernel(const _Float16* __restrict__ qkv,
                      _Float16* __restrict__ attn_out, int p) {
    extern __shared__ float sm[];
    float* Qs = sm;                  // p*64
    float* Ks = Qs + p * HEAD_DIM;   // p*64
    float* Vs = Ks + p * HEAD_DIM;   // p*64
    float* Sc = Vs + p * HEAD_DIM;   // p*p

    const int seq = blockIdx.x;
    const int h   = blockIdx.y;
    const int tid = threadIdx.x;
    const size_t tokBase = (size_t)seq * p;

    for (int i = tid; i < p * HEAD_DIM; i += blockDim.x) {
        const int r = i >> 6, d = i & 63;
        const size_t base = (tokBase + r) * (3 * HID) + h * HEAD_DIM + d;
        Qs[i] = (float)qkv[base];
        Ks[i] = (float)qkv[base + HID];
        Vs[i] = (float)qkv[base + 2 * HID];
    }
    __syncthreads();

    if (tid < p) {
        float mx = -1e30f;
        for (int j = 0; j < p; ++j) {
            float s = 0.0f;
#pragma unroll 8
            for (int d = 0; d < HEAD_DIM; ++d)
                s += Qs[tid * HEAD_DIM + d] * Ks[j * HEAD_DIM + d];
            s *= 0.125f;                      // 1/sqrt(64)
            Sc[tid * p + j] = s;
            mx = fmaxf(mx, s);
        }
        float sum = 0.0f;
        for (int j = 0; j < p; ++j) {
            const float e = __expf(Sc[tid * p + j] - mx);
            Sc[tid * p + j] = e;
            sum += e;
        }
        const float inv = 1.0f / sum;
        for (int d = 0; d < HEAD_DIM; ++d) {
            float o = 0.0f;
            for (int j = 0; j < p; ++j)
                o += Sc[tid * p + j] * Vs[j * HEAD_DIM + d];
            attn_out[(tokBase + tid) * HID + h * HEAD_DIM + d] =
                (_Float16)(o * inv);
        }
    }
}

// ---------------------------------------------------------------------------
// Residual + LayerNorm: x = LN(x + r); writes f32 x and f16 shadow.
// One block per token, 256 threads, 2 elements each.
// ---------------------------------------------------------------------------
__global__ __launch_bounds__(256)
void ln_residual_kernel(const float* __restrict__ x,
                        const float* __restrict__ r,
                        const float* __restrict__ g,
                        const float* __restrict__ b,
                        float* __restrict__ xo,
                        _Float16* __restrict__ xo16) {
    __shared__ float red[256];
    const int t = blockIdx.x;
    const int tid = threadIdx.x;
    const size_t base = (size_t)t * HID;

    float v0 = x[base + tid]       + r[base + tid];
    float v1 = x[base + tid + 256] + r[base + tid + 256];

    red[tid] = v0 + v1;
    __syncthreads();
    for (int s = 128; s > 0; s >>= 1) {
        if (tid < s) red[tid] += red[tid + s];
        __syncthreads();
    }
    const float mean = red[0] * (1.0f / HID);
    __syncthreads();

    const float d0 = v0 - mean, d1 = v1 - mean;
    red[tid] = d0 * d0 + d1 * d1;
    __syncthreads();
    for (int s = 128; s > 0; s >>= 1) {
        if (tid < s) red[tid] += red[tid + s];
        __syncthreads();
    }
    const float inv = rsqrtf(red[0] * (1.0f / HID) + 1e-5f);

    const float o0 = d0 * inv * g[tid]       + b[tid];
    const float o1 = d1 * inv * g[tid + 256] + b[tid + 256];
    xo[base + tid] = o0;         xo16[base + tid] = (_Float16)o0;
    xo[base + tid + 256] = o1;   xo16[base + tid + 256] = (_Float16)o1;
}

// ---------------------------------------------------------------------------
// Final: out[b,o] = mean_{j<10}( dot(acc[b, o*10+j, :]/3, w) + bias )
// One block per output element (16*96), 256 threads.
// ---------------------------------------------------------------------------
__global__ __launch_bounds__(256)
void final_kernel(const float* __restrict__ acc,
                  const float* __restrict__ w,
                  const float* __restrict__ bias,
                  float* __restrict__ out) {
    __shared__ float red[256];
    const int gid = blockIdx.x;
    const int bb = gid / OUT_LEN;
    const int o  = gid % OUT_LEN;
    const int tid = threadIdx.x;

    float s = 0.0f;
    for (int i = tid; i < 10 * HID; i += 256) {
        const int tok = o * 10 + (i >> 9);
        const int c = i & 511;
        s += acc[((size_t)(bb * SS + tok)) * HID + c] * w[c];
    }
    red[tid] = s;
    __syncthreads();
    for (int st = 128; st > 0; st >>= 1) {
        if (tid < st) red[tid] += red[tid + st];
        __syncthreads();
    }
    if (tid == 0) out[gid] = red[0] * (1.0f / 30.0f) + bias[0];
}

// ---------------------------------------------------------------------------
// Host launcher
// ---------------------------------------------------------------------------
extern "C" void kernel_launch(void* const* d_in, const int* in_sizes, int n_in,
                              void* d_out, int out_size, void* d_ws,
                              size_t ws_size, hipStream_t stream) {
    const float* x_in = (const float*)d_in[0];
    const float* ipw  = (const float*)d_in[1];
    const float* ipb  = (const float*)d_in[2];
    const float* wqkv = (const float*)d_in[3];
    const float* bqkv = (const float*)d_in[4];
    const float* wo   = (const float*)d_in[5];
    const float* bo   = (const float*)d_in[6];
    const float* w1   = (const float*)d_in[7];
    const float* b1   = (const float*)d_in[8];
    const float* w2   = (const float*)d_in[9];
    const float* b2   = (const float*)d_in[10];
    const float* g1   = (const float*)d_in[11];
    const float* be1  = (const float*)d_in[12];
    const float* g2   = (const float*)d_in[13];
    const float* be2  = (const float*)d_in[14];
    const float* wout = (const float*)d_in[15];
    const float* bout = (const float*)d_in[16];

    // -------- carve workspace --------
    char* ws = (char*)d_ws;
    size_t off = 0;
    auto alloc = [&](size_t bytes) -> char* {
        char* p = ws + off;
        off = (off + bytes + 255) & ~(size_t)255;
        return p;
    };
    const int NQKV = LAYERS * 3 * HID * HID;   // 4*1536*512
    const int NOUT = LAYERS * HID * HID;       // 4*512*512
    const int NF1  = LAYERS * FFN * HID;       // 4*2048*512
    const int NF2  = LAYERS * HID * FFN;       // 4*512*2048
    const int NACT = TOKENS * HID;             // 15360*512

    _Float16* w16_qkv = (_Float16*)alloc((size_t)NQKV * 2);
    _Float16* w16_out = (_Float16*)alloc((size_t)NOUT * 2);
    _Float16* w16_f1  = (_Float16*)alloc((size_t)NF1 * 2);
    _Float16* w16_f2  = (_Float16*)alloc((size_t)NF2 * 2);
    float*    h0      = (float*)   alloc((size_t)NACT * 4);
    float*    xcur    = (float*)   alloc((size_t)NACT * 4);
    _Float16* x16     = (_Float16*)alloc((size_t)NACT * 2);
    _Float16* qkv16   = (_Float16*)alloc((size_t)TOKENS * 3 * HID * 2);
    _Float16* attn16  = (_Float16*)alloc((size_t)NACT * 2);
    float*    tmp     = (float*)   alloc((size_t)NACT * 4);
    _Float16* mid16   = (_Float16*)alloc((size_t)TOKENS * FFN * 2);
    float*    acc     = (float*)   alloc((size_t)NACT * 4);
    (void)ws_size; (void)in_sizes; (void)n_in; (void)out_size;

    const int CB = 256;
    // -------- convert weights to f16 (once per launch) --------
    convert_f16_kernel<<<2048, CB, 0, stream>>>(wqkv, w16_qkv, NQKV);
    convert_f16_kernel<<<2048, CB, 0, stream>>>(wo,   w16_out, NOUT);
    convert_f16_kernel<<<2048, CB, 0, stream>>>(w1,   w16_f1,  NF1);
    convert_f16_kernel<<<2048, CB, 0, stream>>>(w2,   w16_f2,  NF2);

    // -------- input projection --------
    input_proj_kernel<<<(NACT + CB - 1) / CB, CB, 0, stream>>>(x_in, ipw, ipb, h0);

    const int periods[3] = {24, 48, 96};
    const dim3 gQKV(TOKENS / BM, (3 * HID) / BN);  // 120 x 12
    const dim3 gPRJ(TOKENS / BM, HID / BN);        // 120 x 4
    const dim3 gFF1(TOKENS / BM, FFN / BN);        // 120 x 16
    const dim3 gFF2(TOKENS / BM, HID / BN);        // 120 x 4

    for (int br = 0; br < 3; ++br) {
        const int p = periods[br];
        branch_init_kernel<<<4096, CB, 0, stream>>>(h0, xcur, x16, NACT);

        for (int l = 0; l < LAYERS; ++l) {
            // QKV projection -> f16
            gemm_wmma_kernel<false, true><<<gQKV, 256, 0, stream>>>(
                x16, w16_qkv + (size_t)l * 3 * HID * HID,
                bqkv + l * 3 * HID, (void*)qkv16, TOKENS, 3 * HID, HID);

            // attention
            {
                dim3 ga(TOKENS / p, HEADS);
                size_t smem = (size_t)(3 * p * HEAD_DIM + p * p) * sizeof(float);
                attention_kernel<<<ga, 128, smem, stream>>>(qkv16, attn16, p);
            }

            // output projection -> f32
            gemm_wmma_kernel<false, false><<<gPRJ, 256, 0, stream>>>(
                attn16, w16_out + (size_t)l * HID * HID,
                bo + l * HID, (void*)tmp, TOKENS, HID, HID);

            // x = LN1(x + attn)
            ln_residual_kernel<<<TOKENS, 256, 0, stream>>>(
                xcur, tmp, g1 + l * HID, be1 + l * HID, xcur, x16);

            // FFN1 (+GELU) -> f16
            gemm_wmma_kernel<true, true><<<gFF1, 256, 0, stream>>>(
                x16, w16_f1 + (size_t)l * FFN * HID,
                b1 + l * FFN, (void*)mid16, TOKENS, FFN, HID);

            // FFN2 -> f32
            gemm_wmma_kernel<false, false><<<gFF2, 256, 0, stream>>>(
                mid16, w16_f2 + (size_t)l * HID * FFN,
                b2 + l * HID, (void*)tmp, TOKENS, HID, FFN);

            // x = LN2(x + ffn)
            ln_residual_kernel<<<TOKENS, 256, 0, stream>>>(
                xcur, tmp, g2 + l * HID, be2 + l * HID, xcur, x16);
        }
        accum_kernel<<<4096, CB, 0, stream>>>(xcur, acc, NACT, br == 0 ? 1 : 0);
    }

    // -------- branch mean + output projection + avg pool --------
    final_kernel<<<BB * OUT_LEN, 256, 0, stream>>>(acc, wout, bout, (float*)d_out);
}
